// ConvLayersV2_35089882808651
// MI455X (gfx1250) — matrix operand order repaired and checked by reading, not compiled, not run on hardware
//
#include <hip/hip_runtime.h>

typedef __attribute__((ext_vector_type(16))) _Float16 v16h;
typedef __attribute__((ext_vector_type(8)))  _Float16 v8h;
typedef __attribute__((ext_vector_type(8)))  float    v8f;
typedef __attribute__((ext_vector_type(4)))  float    v4f;

#define IN_CH   3
#define MID_CH  64
#define OUT_CH  16
#define IMG     512
#define OUTD    508

// Per-block tiling: 16x8 output pixels, all 16 output channels.
#define OW 16
#define OH 8
#define HW 18            // h tile width  = OW + 2
#define HH 10            // h tile height = OH + 2
#define HP (HW*HH)       // 180 live h pixels
#define HPP 192          // padded to 12 full N-tiles (no guards anywhere)
#define XW 20            // x tile width  = HW + 2
#define XH 12            // x tile height = HH + 2
#define K2 576           // conv2 reduction = 64ch * 9 taps

#define CAT16(lo, hi) __builtin_shufflevector(lo, hi, 0,1,2,3,4,5,6,7,8,9,10,11,12,13,14,15)

__global__ __launch_bounds__(256)
void conv_fused_wmma(const float* __restrict__ x,
                     const float* __restrict__ w1,
                     const float* __restrict__ w2,
                     float* __restrict__ out)
{
    // LDS layouts chosen so every WMMA fragment is contiguous b128 loads.
    __shared__ __align__(16) _Float16 lds_x   [IN_CH * XH * XW];   //   720 -> 1.4 KB
    __shared__ __align__(16) _Float16 lds_xcol[HPP * 32];          //  6144 -> 12 KB  [pixel][k27pad32]
    __shared__ __align__(16) _Float16 lds_w1p [MID_CH * 32];       //  2048 -> 4 KB   [m][k27pad32]
    __shared__ __align__(16) _Float16 lds_w2t [9 * OUT_CH * 64];   //  9216 -> 18 KB  [tap][m][c]
    __shared__ __align__(16) _Float16 lds_h   [HPP * 64];          // 12288 -> 24 KB  [pixel][c]

    const int tid   = threadIdx.x;
    const int lane  = tid & 31;
    const int wave  = tid >> 5;      // 8 waves
    const int halfs = lane >> 4;     // lane half selects K sub-block
    const int nl    = lane & 15;     // N / M index within 16

    const int tx0 = blockIdx.x * OW;
    const int ty0 = blockIdx.y * OH;
    const int b   = blockIdx.z;

    // ---- Phase 0a: stage x halo (f16), w1 (zero-padded), w2 (transposed) ----
    for (int i = tid; i < IN_CH * XH * XW; i += 256) {
        int c  = i / (XH * XW);
        int r  = i % (XH * XW);
        int yy = r / XW, xx = r % XW;
        int gy = ty0 + yy, gx = tx0 + xx;
        float v = 0.0f;
        if (gy < IMG && gx < IMG)
            v = x[(((size_t)b * IN_CH + c) * IMG + gy) * IMG + gx];
        lds_x[i] = (_Float16)v;
    }
    for (int i = tid; i < MID_CH * 32; i += 256) {       // [m][k], k>=27 zeroed
        int m = i >> 5, k = i & 31;
        lds_w1p[i] = (k < 27) ? (_Float16)w1[m * 27 + k] : (_Float16)0.0f;
    }
    for (int j = tid; j < (OUT_CH * K2) / 4; j += 256) { // coalesced f32x4, scatter to [t][m][c]
        v4f wv = ((const v4f*)w2)[j];
#pragma unroll
        for (int e = 0; e < 4; ++e) {
            int i = 4 * j + e;
            int m = i / K2;
            int r = i - m * K2;
            int c = r / 9;
            int t = r - c * 9;
            lds_w2t[((t * 16 + m) << 6) + c] = (_Float16)wv[e];
        }
    }
    __syncthreads();

    // ---- Phase 0b: im2col of x halo -> [pixel][k] so conv1 B is contiguous ----
    for (int o = tid; o < HPP * 32; o += 256) {
        int p = o >> 5, k = o & 31;
        _Float16 v = (_Float16)0.0f;
        if (p < HP && k < 27) {
            int hy = p / HW, hx = p - hy * HW;
            int c  = k / 9,  t  = k - c * 9;
            v = lds_x[(c * XH + hy + t / 3) * XW + hx + t % 3];
        }
        lds_xcol[o] = v;
    }
    __syncthreads();

    // ---- Phase 1: conv1 (M=64, K=27->32, N=192 padded pixels) ----------------
    // 4 M-tiles x 12 N-tiles = 48 WMMAs over 8 waves; fully uniform, no guards.
#pragma unroll
    for (int it = 0; it < 6; ++it) {
        int pi    = wave + (it << 3);
        int ntile = pi >> 2;
        int mtile = pi & 3;
        int p = (ntile << 4) + nl;           // this lane's pixel (N)
        int m = (mtile << 4) + nl;           // this lane's mid channel (M, A rows)

        const _Float16* ap = lds_w1p + (m << 5) + (halfs << 3);
        v16h a = CAT16(*(const v8h*)ap, *(const v8h*)(ap + 16));

        const _Float16* bp = lds_xcol + (p << 5) + (halfs << 3);
        v16h bb = CAT16(*(const v8h*)bp, *(const v8h*)(bp + 16));

        v8f acc = {};
        acc = __builtin_amdgcn_wmma_f32_16x16x32_f16(
                  false, a, false, bb, (short)0, acc, false, false);

        // 8 accumulator values = 8 consecutive channels -> one b128 store.
        v8h hv;
#pragma unroll
        for (int r = 0; r < 8; ++r) hv[r] = (_Float16)acc[r];
        *(v8h*)(lds_h + (p << 6) + (mtile << 4) + (halfs << 3)) = hv;
    }
    __syncthreads();

    // ---- Phase 2: conv2 as 9 tap-GEMMs, K = 64 channels (2 chunks of 32) ----
    {
        int oy = wave;                       // one output row per wave (OW==16)
        int ox = nl;
        v8f acc = {};
#pragma unroll
        for (int t = 0; t < 9; ++t) {
            int ty = t / 3, tx = t % 3;                  // constants after unroll
            int prow = (oy + ty) * HW + (ox + tx);       // h pixel for this lane
#pragma unroll
            for (int cc = 0; cc < 2; ++cc) {
                const _Float16* ap = lds_w2t + ((t * 16 + nl) << 6)
                                             + (cc << 5) + (halfs << 3);
                v16h a = CAT16(*(const v8h*)ap, *(const v8h*)(ap + 16));

                const _Float16* bp = lds_h + (prow << 6)
                                           + (cc << 5) + (halfs << 3);
                v16h bb = CAT16(*(const v8h*)bp, *(const v8h*)(bp + 16));

                acc = __builtin_amdgcn_wmma_f32_16x16x32_f16(
                          false, a, false, bb, (short)0, acc, false, false);
            }
        }
        // Store C fragment: VGPR r -> out channel r (+8 for upper lane half).
        int oyg = ty0 + oy, oxg = tx0 + ox;
        if (oyg < OUTD && oxg < OUTD) {
#pragma unroll
            for (int r = 0; r < 8; ++r) {
                int oc = r + (halfs << 3);
                out[(((size_t)b * OUT_CH + oc) * OUTD + oyg) * OUTD + oxg] = acc[r];
            }
        }
    }
}

extern "C" void kernel_launch(void* const* d_in, const int* in_sizes, int n_in,
                              void* d_out, int out_size, void* d_ws, size_t ws_size,
                              hipStream_t stream) {
    const float* x  = (const float*)d_in[0];
    const float* w1 = (const float*)d_in[1];
    const float* w2 = (const float*)d_in[2];
    float* out = (float*)d_out;

    dim3 grid((OUTD + OW - 1) / OW,   // 32
              (OUTD + OH - 1) / OH,   // 64
              8);                     // batch
    conv_fused_wmma<<<grid, dim3(256), 0, stream>>>(x, w1, w2, out);
}
